// RelativePositionAttention_86320252715348
// MI455X (gfx1250) — compile-verified, matrix operand order checked by reading, not run on hardware
//
#include <hip/hip_runtime.h>
#include <hip/hip_bf16.h>

// ---------------- CDNA5 WMMA types & helpers (wave32, gfx1250) ----------------
typedef __attribute__((ext_vector_type(16))) __bf16 v16bf;
typedef __attribute__((ext_vector_type(8)))  float  v8f;
typedef __attribute__((ext_vector_type(4)))  unsigned int v4u;
typedef __attribute__((ext_vector_type(8)))  int v8i;
typedef __attribute__((ext_vector_type(4)))  int v4i;

static __device__ __forceinline__ unsigned short f2bfu(float f) {
  unsigned u = __builtin_bit_cast(unsigned, f);
  unsigned r = u + 0x7FFFu + ((u >> 16) & 1u);   // round-to-nearest-even
  return (unsigned short)(r >> 16);
}
static __device__ __forceinline__ float bfu2f(unsigned short h) {
  unsigned u = ((unsigned)h) << 16;
  return __builtin_bit_cast(float, u);
}
static __device__ __forceinline__ __bf16 u2bf(unsigned short h) {
  return __builtin_bit_cast(__bf16, h);
}

// A-fragment / B-from-transposed: lane gives row (M or N), elem i gives K.
// K map per ISA 16-bit A 16x32 layout: k = (i&7) + 8*(lane>=16) + 16*(i>=8)
static __device__ __forceinline__ v16bf load_frag_rm(const unsigned short* base, int stride) {
  int lane = threadIdx.x & 31, half = lane >> 4, m = lane & 15;
  v16bf f;
#pragma unroll
  for (int i = 0; i < 16; ++i) {
    int k = (i & 7) + (half << 3) + ((i >> 3) << 4);
    f[i] = u2bf(base[m * stride + k]);
  }
  return f;
}
// B-fragment from row-major KxN storage: element base[k*stride + n]
static __device__ __forceinline__ v16bf load_frag_cm(const unsigned short* base, int stride) {
  int lane = threadIdx.x & 31, half = lane >> 4, n = lane & 15;
  v16bf f;
#pragma unroll
  for (int i = 0; i < 16; ++i) {
    int k = (i & 7) + (half << 3) + ((i >> 3) << 4);
    f[i] = u2bf(base[k * stride + n]);
  }
  return f;
}
// A-fragment from fp32 LDS (attn probabilities), convert to bf16 on the fly
static __device__ __forceinline__ v16bf load_frag_rm_f32(const float* base, int stride) {
  int lane = threadIdx.x & 31, half = lane >> 4, m = lane & 15;
  v16bf f;
#pragma unroll
  for (int i = 0; i < 16; ++i) {
    int k = (i & 7) + (half << 3) + ((i >> 3) << 4);
    f[i] = u2bf(f2bfu(base[m * stride + k]));
  }
  return f;
}
static __device__ __forceinline__ v8f wmma_bf16(v16bf a, v16bf b, v8f c) {
  return __builtin_amdgcn_wmma_f32_16x16x32_bf16(false, a, false, b, (short)0, c, false, false);
}

// ---- Tensor Data Mover: DMA a 2D bf16 tile (dim0 x dim1, row stride stride0
// elements) from global memory into LDS at byte offset lds_off. D# per ISA
// cdna5 §8.3/8.4: group0 = {flags, lds_addr, global_addr, type}, group1 packs
// data_size / tensor dims / tile dims / strides. 6-arg (clang-23) builtin form:
// (v4u g0, v8i g1, v4i g2, v4i g3, v8i extra, i32 cpol). 2-D tile -> g2/g3 unused.
static __device__ __forceinline__ void tdm_load_2d_bf16(
    unsigned lds_off, const void* gptr, int dim0, int dim1, int stride0) {
  unsigned long long ga = (unsigned long long)(__SIZE_TYPE__)gptr;
  v4u g0;
  g0[0] = 1u;                                        // count=1, user descriptor
  g0[1] = lds_off;                                   // lds_addr (bytes)
  g0[2] = (unsigned)(ga & 0xFFFFFFFFu);              // global_addr[31:0]
  g0[3] = (unsigned)((ga >> 32) & 0x01FFFFFFu) | (2u << 30);  // addr[56:32] | type=2
  v8i g1;
  g1[0] = (1 << 16);                                 // data_size=1 (2 bytes)
  g1[1] = (dim0 & 0xFFFF) << 16;                     // tensor_dim0[15:0]
  g1[2] = ((unsigned)dim0 >> 16) | ((dim1 & 0xFFFF) << 16);   // dim0 hi | dim1 lo
  g1[3] = ((unsigned)dim1 >> 16) | ((dim0 & 0xFFFF) << 16);   // dim1 hi | tile_dim0
  g1[4] = (dim1 & 0xFFFF);                           // tile_dim1 (tile_dim2=0 -> 2D)
  g1[5] = stride0;                                   // tensor_dim0_stride[31:0]
  g1[6] = 0;
  g1[7] = 0;
  v4i z4 = {};
  v8i z8 = {};
  __builtin_amdgcn_tensor_load_to_lds(g0, g1, z4, z4, z8, 0);
}

#define HID 1024
#define NH 16
#define HD 64
#define SQ 1024
#define NB 2
#define MAXREL 32
#define NP 65

// ---------------- Kernel 1: QKV projection (fp32 in -> bf16 [B,H,S,64]) ------
__global__ __launch_bounds__(128) void k_qkv_proj(
    const float* __restrict__ q_in, const float* __restrict__ k_in, const float* __restrict__ v_in,
    const float* __restrict__ Wq, const float* __restrict__ bq,
    const float* __restrict__ Wk, const float* __restrict__ bk,
    const float* __restrict__ Wv, const float* __restrict__ bv,
    unsigned short* __restrict__ qw, unsigned short* __restrict__ kw, unsigned short* __restrict__ vw) {
  const int which = blockIdx.z;
  const float* X    = which == 0 ? q_in : which == 1 ? k_in : v_in;
  const float* W    = which == 0 ? Wq   : which == 1 ? Wk   : Wv;
  const float* bias = which == 0 ? bq   : which == 1 ? bk   : bv;
  unsigned short* out = which == 0 ? qw : which == 1 ? kw : vw;

  __shared__ unsigned short At[64 * 32];   // A tile 64(M) x 32(K)
  __shared__ unsigned short Bt[32 * 64];   // B tile 32(K) x 64(N)

  const int tid = threadIdx.x, wave = tid >> 5;
  const int n0 = blockIdx.x * 64, m0 = blockIdx.y * 64;

  v8f acc[4] = {};
  for (int k0 = 0; k0 < HID; k0 += 32) {
#pragma unroll
    for (int e = 0; e < 16; ++e) {            // A: coalesced over K
      int idx = tid + 128 * e;
      int r = idx >> 5, c = idx & 31;
      At[idx] = f2bfu(X[(m0 + r) * HID + k0 + c]);
    }
#pragma unroll
    for (int e = 0; e < 16; ++e) {            // B[k][n] = W[n0+n][k0+k], coalesced over k
      int idx = tid + 128 * e;
      int n = idx >> 5, kk = idx & 31;
      Bt[kk * 64 + n] = f2bfu(W[(n0 + n) * HID + k0 + kk]);
    }
    __syncthreads();
    v16bf a = load_frag_rm(&At[(wave * 16) * 32], 32);
#pragma unroll
    for (int j = 0; j < 4; ++j) {
      v16bf b = load_frag_cm(&Bt[j * 16], 64);
      acc[j] = wmma_bf16(a, b, acc[j]);
    }
    __syncthreads();
  }
  const int lane = tid & 31, half = lane >> 4, nl = lane & 15;
#pragma unroll
  for (int j = 0; j < 4; ++j) {
#pragma unroll
    for (int r = 0; r < 8; ++r) {
      int m = m0 + wave * 16 + r + 8 * half;  // global row = b*S + s
      int n = n0 + j * 16 + nl;               // hidden col = h*64 + d
      float val = acc[j][r] + bias[n];
      int b = m >> 10, s = m & 1023, h = n >> 6, d = n & 63;
      out[(((b * NH + h) << 10) | s) * HD + d] = f2bfu(val);
    }
  }
}

// ---------------- Kernel 2: qrel[bh][i][p] = q[bh][i][:] . rel_k[p][:] -------
__global__ __launch_bounds__(256) void k_qrel(
    const unsigned short* __restrict__ qw, const float* __restrict__ relk,
    float* __restrict__ qrel) {
  __shared__ float rk[NP * HD];
  for (int e = threadIdx.x; e < NP * HD; e += 256) rk[e] = relk[e];
  __syncthreads();
  int row = blockIdx.x * 256 + threadIdx.x;      // 0 .. B*H*S-1
  const unsigned short* qr = qw + row * HD;
  float q[HD];
#pragma unroll
  for (int d = 0; d < HD; ++d) q[d] = bfu2f(qr[d]);
  float* o = qrel + row * NP;
  for (int p = 0; p < NP; ++p) {
    float sum = 0.f;
#pragma unroll
    for (int d = 0; d < HD; ++d) sum += q[d] * rk[p * HD + d];
    o[p] = sum;
  }
}

// ---------------- Kernel 3: fused attention for one (bh, 32-row block) -------
// Dynamic LDS: scores [32][1028] f32 + attnp [32][65] f32 + K/V slab [1024][64] bf16
__global__ __launch_bounds__(256) void k_attn(
    const unsigned short* __restrict__ qw, const unsigned short* __restrict__ kw,
    const unsigned short* __restrict__ vw, const float* __restrict__ qrel,
    const unsigned char* __restrict__ mask, const float* __restrict__ relv,
    unsigned short* __restrict__ attended, float* __restrict__ meanOut) {
  extern __shared__ float smem[];
  const int SST = 1028;
  float* scores = smem;                               // 32*1028 f32
  float* attnp  = smem + 32 * SST;                    // 32*65  f32
  unsigned short* kvbuf = (unsigned short*)(attnp + 32 * NP);  // 1024*64 bf16
  const unsigned kv_ldsoff =
      __builtin_amdgcn_groupstaticsize() + (unsigned)((32 * SST + 32 * NP) * 4);

  const int bh = blockIdx.y, b = bh >> 4, h = bh & 15;
  const int i0 = blockIdx.x * 32;
  const int tid = threadIdx.x, wave = tid >> 5;
  const int lane = tid & 31, half = lane >> 4, nl = lane & 15;
  const float scale = 0.125f;  // 64^-0.5

  // Stage the whole per-head K slab [1024 x 64] bf16 into LDS via the TDM.
  if (wave == 0) {
    tdm_load_2d_bf16(kv_ldsoff, kw + (size_t)(bh << 10) * HD, HD, SQ, HD);
    __builtin_amdgcn_s_wait_tensorcnt(0);
  }
  __syncthreads();

  // ---- Phase A: content scores via WMMA + relative term + scale + mask ----
  {
    const int it = wave >> 2;       // which 16-row i-tile (0/1)
    const int jcw = wave & 3;       // which 16-col chunk within a 64-col step
    const unsigned short* qbase = qw + ((bh << 10) + i0 + it * 16) * HD;
    v16bf a0 = load_frag_rm(qbase, HD);        // d = 0..31
    v16bf a1 = load_frag_rm(qbase + 32, HD);   // d = 32..63
    for (int jt = 0; jt < 16; ++jt) {
      int j0 = jt * 64 + jcw * 16;
      const unsigned short* kbase = kvbuf + j0 * HD;  // K^T frag: base[n*64+k]
      v16bf b0 = load_frag_rm(kbase, HD);
      v16bf b1 = load_frag_rm(kbase + 32, HD);
      v8f c = {};
      c = wmma_bf16(a0, b0, c);
      c = wmma_bf16(a1, b1, c);
#pragma unroll
      for (int r = 0; r < 8; ++r) {
        int il = it * 16 + r + 8 * half;
        int i = i0 + il, j = j0 + nl;
        int delta = i - j;
        int p = (delta > MAXREL ? MAXREL : (delta < -MAXREL ? -MAXREL : delta)) + MAXREL;
        float s = (c[r] + qrel[((bh << 10) + i) * NP + p]) * scale;
        if (!mask[(b << 10) + j]) s = -1e9f;
        scores[il * SST + j] = s;
      }
    }
  }
  __syncthreads();   // all K reads done -> kvbuf reusable

  // Kick off the V slab DMA; it overlaps the softmax phase below.
  if (wave == 0) {
    tdm_load_2d_bf16(kv_ldsoff, vw + (size_t)(bh << 10) * HD, HD, SQ, HD);
  }

  // ---- Phase B: exact softmax (8 lanes per row) + mean accumulation ----
  {
    int row = tid >> 3, sub = tid & 7;
    float* sr = scores + row * SST;
    float mx = -3.4e38f;
    for (int j = sub; j < SQ; j += 8) mx = fmaxf(mx, sr[j]);
#pragma unroll
    for (int o = 4; o >= 1; o >>= 1) mx = fmaxf(mx, __shfl_xor(mx, o, 8));
    float sum = 0.f;
    for (int j = sub; j < SQ; j += 8) { float e = __expf(sr[j] - mx); sr[j] = e; sum += e; }
#pragma unroll
    for (int o = 4; o >= 1; o >>= 1) sum += __shfl_xor(sum, o, 8);
    float inv = 1.0f / sum;
    float* mrow = meanOut + ((b << 10) + i0 + row) * SQ;
    for (int j = sub; j < SQ; j += 8) {
      float a = sr[j] * inv;
      sr[j] = a;
      atomicAdd(&mrow[j], a * 0.0625f);   // /NUM_HEADS
    }
  }
  __syncthreads();

  // ---- attnp: 65-bin histogram of attn over relative position ----
  for (int e = tid; e < 32 * NP; e += 256) {
    int il = e / NP, p = e % NP;
    int i = i0 + il;
    const float* sr = scores + il * SST;
    float v = 0.f;
    if (p == 0) {                       // delta <= -32  -> j >= i+32
      for (int j = i + MAXREL; j < SQ; ++j) v += sr[j];
    } else if (p == NP - 1) {           // delta >= 32  -> j <= i-32
      int jmax = i - MAXREL;
      for (int j = 0; j <= jmax; ++j) v += sr[j];
    } else {                            // j = i + 32 - p
      int j = i + MAXREL - p;
      if (j >= 0 && j < SQ) v = sr[j];
    }
    attnp[e] = v;
  }
  if (wave == 0) __builtin_amdgcn_s_wait_tensorcnt(0);  // V slab landed
  __syncthreads();

  // ---- Phase C: attended = attn @ V (WMMA, V from LDS) + attnp @ rel_v ----
  {
    const int it = wave >> 2;
    const int d0 = (wave & 3) * 16;
    v8f acc = {};
    for (int j0 = 0; j0 < SQ; j0 += 32) {
      v16bf a = load_frag_rm_f32(scores + (it * 16) * SST + j0, SST);
      v16bf bf = load_frag_cm(kvbuf + j0 * HD + d0, HD);  // V[j][d]
      acc = wmma_bf16(a, bf, acc);
    }
#pragma unroll
    for (int r = 0; r < 8; ++r) {
      int il = it * 16 + r + 8 * half;
      int i = i0 + il, d = d0 + nl;
      float val = acc[r];
      const float* ap = attnp + il * NP;
      for (int p = 0; p < NP; ++p) val += ap[p] * relv[p * HD + d];
      attended[((b << 10) + i) * HID + h * HD + d] = f2bfu(val);
    }
  }
}

// ---------------- Kernel 4: output projection (bf16 A, fp32 W -> fp32 out) ---
__global__ __launch_bounds__(128) void k_out_proj(
    const unsigned short* __restrict__ A, const float* __restrict__ W,
    const float* __restrict__ bias, float* __restrict__ out) {
  __shared__ unsigned short At[64 * 32];
  __shared__ unsigned short Bt[32 * 64];
  const int tid = threadIdx.x, wave = tid >> 5;
  const int n0 = blockIdx.x * 64, m0 = blockIdx.y * 64;

  v8f acc[4] = {};
  for (int k0 = 0; k0 < HID; k0 += 32) {
#pragma unroll
    for (int e = 0; e < 16; ++e) {
      int idx = tid + 128 * e;
      int r = idx >> 5, c = idx & 31;
      At[idx] = A[(m0 + r) * HID + k0 + c];
    }
#pragma unroll
    for (int e = 0; e < 16; ++e) {
      int idx = tid + 128 * e;
      int n = idx >> 5, kk = idx & 31;
      Bt[kk * 64 + n] = f2bfu(W[(n0 + n) * HID + k0 + kk]);
    }
    __syncthreads();
    v16bf a = load_frag_rm(&At[(wave * 16) * 32], 32);
#pragma unroll
    for (int j = 0; j < 4; ++j) {
      v16bf b = load_frag_cm(&Bt[j * 16], 64);
      acc[j] = wmma_bf16(a, b, acc[j]);
    }
    __syncthreads();
  }
  const int lane = tid & 31, half = lane >> 4, nl = lane & 15;
#pragma unroll
  for (int j = 0; j < 4; ++j) {
#pragma unroll
    for (int r = 0; r < 8; ++r) {
      int m = m0 + wave * 16 + r + 8 * half;
      int n = n0 + j * 16 + nl;
      out[m * HID + n] = acc[j][r] + bias[n];
    }
  }
}

// ---------------- Host launcher ----------------
extern "C" void kernel_launch(void* const* d_in, const int* in_sizes, int n_in,
                              void* d_out, int out_size, void* d_ws, size_t ws_size,
                              hipStream_t stream) {
  const float* query = (const float*)d_in[0];
  const float* key_  = (const float*)d_in[1];
  const float* value = (const float*)d_in[2];
  const unsigned char* mask = (const unsigned char*)d_in[3];
  const float* Wq = (const float*)d_in[4];  const float* bq = (const float*)d_in[5];
  const float* Wk = (const float*)d_in[6];  const float* bk = (const float*)d_in[7];
  const float* Wv = (const float*)d_in[8];  const float* bv = (const float*)d_in[9];
  const float* Wo = (const float*)d_in[10]; const float* bo = (const float*)d_in[11];
  const float* relk = (const float*)d_in[12];
  const float* relv = (const float*)d_in[13];

  char* ws = (char*)d_ws;
  unsigned short* qw   = (unsigned short*)(ws + 0);
  unsigned short* kw   = (unsigned short*)(ws + (4u << 20));
  unsigned short* vw   = (unsigned short*)(ws + (8u << 20));
  float*          qrel = (float*)(ws + (12u << 20));
  unsigned short* attw = (unsigned short*)(ws + (12u << 20) + 8519680u);  // 32*1024*65*4

  float* out  = (float*)d_out;
  float* mean = out + (size_t)NB * SQ * HID;  // second output: attn.mean over heads

  (void)hipMemsetAsync(mean, 0, (size_t)NB * SQ * SQ * sizeof(float), stream);

  // 1. QKV projections
  k_qkv_proj<<<dim3(HID / 64, NB * SQ / 64, 3), dim3(128), 0, stream>>>(
      query, key_, value, Wq, bq, Wk, bk, Wv, bv, qw, kw, vw);

  // 2. q . rel_k table
  k_qrel<<<dim3(NB * NH * SQ / 256), dim3(256), 0, stream>>>(qw, relk, qrel);

  // 3. fused attention: scores slab + attnp + 128KB K/V TDM buffer in LDS
  const size_t smem = (size_t)(32 * 1028 + 32 * NP) * sizeof(float)
                    + (size_t)SQ * HD * sizeof(unsigned short);   // 270,976 B
  (void)hipFuncSetAttribute(reinterpret_cast<const void*>(k_attn),
                            hipFuncAttributeMaxDynamicSharedMemorySize, (int)smem);
  k_attn<<<dim3(SQ / 32, NB * NH), dim3(256), smem, stream>>>(
      qw, kw, vw, qrel, mask, relv, attw, mean);

  // 4. output projection
  k_out_proj<<<dim3(HID / 64, NB * SQ / 64), dim3(128), 0, stream>>>(attw, Wo, bo, out);
}
